// att_cell_22479858827484
// MI455X (gfx1250) — matrix operand order repaired and checked by reading, not compiled, not run on hardware
//
#include <hip/hip_runtime.h>
#include <math.h>

// MI455X / gfx1250, wave32.
// Pipeline: pack f16 operands -> attention/agg (f32 VALU, agg stored f16) ->
// double-buffered 128x128x64 LDS-tiled GEMM with V_WMMA_F32_16X16X32_F16
// (f16 inputs, f32 accumulation) for spatial gates and LSTM gates -> LSTM
// elementwise. KT=64 keeps the prefetch pipeline register-resident (no
// scratch spills) and LDS at 72KB -> 4 blocks/WGP.

typedef _Float16 h8  __attribute__((ext_vector_type(8)));
typedef _Float16 h16 __attribute__((ext_vector_type(16)));
typedef float    v8f __attribute__((ext_vector_type(8)));

#define Nn      8192    // B*S
#define INc     128
#define Hc      512
#define NHEADc  4
#define NNc     8
#define AGGK    2048    // NHEAD*H
#define G4      2048    // 4*H
#define KC      640     // IN + H
#define AROWS   16      // rows per attention block

#define MT      128     // GEMM M tile
#define NT      128     // GEMM N tile
#define KT      64      // GEMM K stage
#define PSTR    72      // LDS panel stride in halves (64+8): 144B rows, 16B aligned

__device__ __forceinline__ float sigmoidf_(float v) { return 1.0f / (1.0f + __expf(-v)); }

// ---------------------------------------------------------------------------
// operand packing (f32 -> f16)
// ---------------------------------------------------------------------------
__global__ __launch_bounds__(256) void cvt_f16_kernel(const float* __restrict__ s,
                                                      _Float16* __restrict__ d, int n) {
    int i = blockIdx.x * 256 + threadIdx.x;
    if (i < n) d[i] = (_Float16)s[i];
}

// WgH[col*640 + k] = k<128 ? w_ih[col,k] : w_hh[col,k-128]
__global__ __launch_bounds__(256) void pack_wg_kernel(const float* __restrict__ wih,
                                                      const float* __restrict__ whh,
                                                      _Float16* __restrict__ wg) {
    int i = blockIdx.x * 256 + threadIdx.x;
    if (i >= G4 * KC) return;
    int col = i / KC, k = i - col * KC;
    float v = (k < INc) ? wih[col * INc + k] : whh[col * Hc + (k - INc)];
    wg[i] = (_Float16)v;
}

// AgH[n*640 + k] = x[n,k]  (k<128); h' part (k>=128) filled by spatial epilogue
__global__ __launch_bounds__(256) void pack_x_kernel(const float* __restrict__ x,
                                                     _Float16* __restrict__ Ag) {
    int i = blockIdx.x * 256 + threadIdx.x;
    if (i >= Nn * INc) return;
    int n = i >> 7, k = i & (INc - 1);
    Ag[(size_t)n * KC + k] = (_Float16)x[i];
}

// ---------------------------------------------------------------------------
// Kernel A: GAT attention (logits, leaky-relu, softmax) + aggregation.
// One block = 16 rows, 8 waves. agg written to global as f16 (GEMM operand).
// hs is read twice; 128MB < 192MB L2 so the second pass is L2-resident.
// ---------------------------------------------------------------------------
__global__ __launch_bounds__(256)
void att_agg_kernel(const float* __restrict__ h_in,
                    const float* __restrict__ hs,
                    const float* __restrict__ a_src,
                    _Float16* __restrict__ aggH)
{
    __shared__ float wgtS[AROWS * NNc * NHEADc];   // 512
    __shared__ float selfS[AROWS * NHEADc];        // 64

    const int tid  = threadIdx.x;
    const int lane = tid & 31;
    const int wave = tid >> 5;
    const int n0   = blockIdx.x * AROWS;

    // logits: each wave handles 2 rows
    for (int rl = 0; rl < 2; ++rl) {
        const int rr = wave * 2 + rl;
        const int n  = n0 + rr;
        float a0 = 0.f, a1 = 0.f, a2 = 0.f, a3 = 0.f;
        #pragma unroll 4
        for (int i = 0; i < 16; ++i) {
            int hh = lane + 32 * i;
            float hv = h_in[(size_t)n * Hc + hh];
            const float* ap = a_src + (size_t)hh * NHEADc;
            a0 += hv * ap[0]; a1 += hv * ap[1]; a2 += hv * ap[2]; a3 += hv * ap[3];
        }
        for (int m = 16; m >= 1; m >>= 1) {
            a0 += __shfl_xor(a0, m, 32); a1 += __shfl_xor(a1, m, 32);
            a2 += __shfl_xor(a2, m, 32); a3 += __shfl_xor(a3, m, 32);
        }
        if (lane == 0) {
            selfS[rr*4+0] = a0; selfS[rr*4+1] = a1;
            selfS[rr*4+2] = a2; selfS[rr*4+3] = a3;
        }
        for (int k = 0; k < NNc; ++k) {
            float b0 = 0.f, b1 = 0.f, b2 = 0.f, b3 = 0.f;
            const float* hsp = hs + ((size_t)n * NNc + k) * Hc;
            #pragma unroll 4
            for (int i = 0; i < 16; ++i) {
                int hh = lane + 32 * i;
                float hv = hsp[hh];
                const float* ap = a_src + (size_t)(Hc + hh) * NHEADc;
                b0 += hv * ap[0]; b1 += hv * ap[1]; b2 += hv * ap[2]; b3 += hv * ap[3];
            }
            for (int m = 16; m >= 1; m >>= 1) {
                b0 += __shfl_xor(b0, m, 32); b1 += __shfl_xor(b1, m, 32);
                b2 += __shfl_xor(b2, m, 32); b3 += __shfl_xor(b3, m, 32);
            }
            if (lane == 0) {
                wgtS[(rr*NNc+k)*4+0] = b0; wgtS[(rr*NNc+k)*4+1] = b1;
                wgtS[(rr*NNc+k)*4+2] = b2; wgtS[(rr*NNc+k)*4+3] = b3;
            }
        }
    }
    __syncthreads();

    // leaky-relu + softmax over k, one thread per (row, head)
    if (tid < AROWS * NHEADc) {
        int rr = tid >> 2, m = tid & 3;
        float sv = selfS[rr*4+m];
        float e[NNc]; float mx = -3.0e38f;
        #pragma unroll
        for (int k = 0; k < NNc; ++k) {
            float v = sv + wgtS[(rr*NNc+k)*4+m];
            v = (v >= 0.f) ? v : 0.2f * v;
            e[k] = v; mx = fmaxf(mx, v);
        }
        float s = 0.f;
        #pragma unroll
        for (int k = 0; k < NNc; ++k) { e[k] = __expf(e[k] - mx); s += e[k]; }
        float inv = 1.0f / s;
        #pragma unroll
        for (int k = 0; k < NNc; ++k) wgtS[(rr*NNc+k)*4+m] = e[k] * inv;
    }
    __syncthreads();

    // aggregation -> aggH (f16)
    for (int rr = 0; rr < AROWS; ++rr) {
        const int n = n0 + rr;
        const float* hsp = hs + (size_t)n * NNc * Hc;
        for (int hh = tid; hh < Hc; hh += 256) {
            float hv[NNc];
            #pragma unroll
            for (int k = 0; k < NNc; ++k) hv[k] = hsp[(size_t)k * Hc + hh];
            #pragma unroll
            for (int m = 0; m < NHEADc; ++m) {
                float s = 0.f;
                #pragma unroll
                for (int k = 0; k < NNc; ++k) s += wgtS[(rr*NNc+k)*4+m] * hv[k];
                aggH[(size_t)n * AGGK + m * Hc + hh] = (_Float16)s;
            }
        }
    }
}

// ---------------------------------------------------------------------------
// Generic 128x128x64 tiled f16 GEMM, f32 accumulation, double-buffered LDS
// panels, V_WMMA_F32_16X16X32_F16. 256 threads = 8 waves (4 M-groups x
// 2 N-groups); each wave owns a 32x64 region = 2x4 WMMA tiles.
// MODE 0: spatial-gate epilogue (sigmoid, write c' f32 and h' f16 into AgH)
// MODE 1: LSTM-gate epilogue (add biases, write gates f32)
// ---------------------------------------------------------------------------
__device__ __forceinline__ void fetch_panel_regs(int4* r, const _Float16* __restrict__ g,
                                                 int ld, int k0, int tid) {
    #pragma unroll
    for (int i = 0; i < 4; ++i) {
        int cid = tid + i * 256;          // 1024 chunks of 16B (8 halves)
        int row = cid >> 3, pos = cid & 7;
        r[i] = *(const int4*)(g + (size_t)row * ld + k0 + pos * 8);
    }
}
__device__ __forceinline__ void store_panel_regs(_Float16* __restrict__ s,
                                                 const int4* r, int tid) {
    #pragma unroll
    for (int i = 0; i < 4; ++i) {
        int cid = tid + i * 256;
        int row = cid >> 3, pos = cid & 7;
        *(int4*)(s + row * PSTR + pos * 8) = r[i];
    }
}

// A frag (16x32): lane m=lane%16, half=lane/16; halves [kk+half*8, +8) then
// [kk+16+half*8, +8) -> two ds_load_b128, matches ISA A layout.
__device__ __forceinline__ h16 load_fragA(const _Float16* __restrict__ As,
                                          int mt, int kk, int lane) {
    int m = lane & 15, half = lane >> 4;
    const _Float16* p = As + (mt * 16 + m) * PSTR + kk + half * 8;
    h8 lo = *(const h8*)p;
    h8 hi = *(const h8*)(p + 16);
    return __builtin_shufflevector(lo, hi, 0,1,2,3,4,5,6,7,8,9,10,11,12,13,14,15);
}
// B frag (32x16): lane n=lane%16; halves [kk+half*8, +8) then [kk+16+half*8, +8).
__device__ __forceinline__ h16 load_fragB(const _Float16* __restrict__ Bs,
                                          int nt, int kk, int lane) {
    int n = lane & 15, half = lane >> 4;
    const _Float16* p = Bs + (nt * 16 + n) * PSTR + kk + half * 8;
    h8 lo = *(const h8*)p;
    h8 hi = *(const h8*)(p + 16);
    return __builtin_shufflevector(lo, hi, 0,1,2,3,4,5,6,7,8,9,10,11,12,13,14,15);
}

template <int MODE>
__global__ __launch_bounds__(256)
void gemm_f16_kernel(const _Float16* __restrict__ A, int lda,
                     const _Float16* __restrict__ B, int ldb, int Ktot,
                     const float* __restrict__ bias0,   // b_fg / b_ih
                     const float* __restrict__ bias1,   // b_ig / b_hh
                     const float* __restrict__ e_h,     // mode0: h
                     const float* __restrict__ e_c,     // mode0: c
                     float* __restrict__ e_cprime,      // mode0
                     _Float16* __restrict__ e_AgH,      // mode0: h' dest
                     float* __restrict__ e_gates)      // mode1
{
    extern __shared__ _Float16 sm[];
    _Float16* AsB = sm;                   // 2 * MT*PSTR
    _Float16* BsB = sm + 2 * MT * PSTR;   // 2 * NT*PSTR

    const int tid = threadIdx.x, lane = tid & 31, wave = tid >> 5;
    const int m0 = blockIdx.x * MT, n0 = blockIdx.y * NT;
    const int mw = wave & 3, nw = wave >> 2;
    const _Float16* Ag = A + (size_t)m0 * lda;
    const _Float16* Bg = B + (size_t)n0 * ldb;
    const int nstage = Ktot / KT;

    {   // stage 0
        int4 r[4];
        fetch_panel_regs(r, Ag, lda, 0, tid); store_panel_regs(AsB, r, tid);
        fetch_panel_regs(r, Bg, ldb, 0, tid); store_panel_regs(BsB, r, tid);
    }
    __syncthreads();

    v8f acc[2][4] = {};
    int buf = 0;
    #pragma unroll 1
    for (int s = 0; s < nstage; ++s) {
        // unconditional prefetch; last stage re-fetches itself into the dead
        // buffer (harmless) -> no control flow around the register pipeline.
        const int kNext = (s + 1 < nstage) ? (s + 1) * KT : s * KT;
        int4 pa[4], pb[4];
        fetch_panel_regs(pa, Ag, lda, kNext, tid);
        fetch_panel_regs(pb, Bg, ldb, kNext, tid);

        const _Float16* As = AsB + buf * (MT * PSTR);
        const _Float16* Bs = BsB + buf * (NT * PSTR);
        #pragma unroll
        for (int kk = 0; kk < KT; kk += 32) {
            h16 af0 = load_fragA(As, 2 * mw + 0, kk, lane);
            h16 af1 = load_fragA(As, 2 * mw + 1, kk, lane);
            h16 bf[4];
            #pragma unroll
            for (int j = 0; j < 4; ++j) bf[j] = load_fragB(Bs, 4 * nw + j, kk, lane);
            #pragma unroll
            for (int j = 0; j < 4; ++j) {
                acc[0][j] = __builtin_amdgcn_wmma_f32_16x16x32_f16(
                    false, af0, false, bf[j], (short)0, acc[0][j], false, false);
                acc[1][j] = __builtin_amdgcn_wmma_f32_16x16x32_f16(
                    false, af1, false, bf[j], (short)0, acc[1][j], false, false);
            }
        }
        store_panel_regs(AsB + (buf ^ 1) * (MT * PSTR), pa, tid);
        store_panel_regs(BsB + (buf ^ 1) * (NT * PSTR), pb, tid);
        __syncthreads();
        buf ^= 1;
    }

    // epilogue (C/D layout: VGPR e -> M = e + half*8, N = lane%16)
    const int n_ = lane & 15, half = lane >> 4;
    #pragma unroll
    for (int i = 0; i < 2; ++i) {
        #pragma unroll
        for (int j = 0; j < 4; ++j) {
            int colg = n0 + (4 * nw + j) * 16 + n_;
            #pragma unroll
            for (int e = 0; e < 8; ++e) {
                int row = m0 + (2 * mw + i) * 16 + half * 8 + e;
                float v = acc[i][j][e];
                if (MODE == 0) {
                    if (colg < Hc) {           // forget gate -> c'
                        float g = sigmoidf_(v + bias0[colg]);
                        e_cprime[(size_t)row * Hc + colg] =
                            e_c[(size_t)row * Hc + colg] * g;
                    } else {                   // input gate -> h' (f16 into AgH)
                        int cc = colg - Hc;
                        float g = sigmoidf_(v + bias1[cc]);
                        e_AgH[(size_t)row * KC + INc + cc] =
                            (_Float16)(e_h[(size_t)row * Hc + cc] * g);
                    }
                } else {
                    e_gates[(size_t)row * G4 + colg] = v + bias0[colg] + bias1[colg];
                }
            }
        }
    }
}

// ---------------------------------------------------------------------------
// LSTM elementwise combine
// ---------------------------------------------------------------------------
__global__ __launch_bounds__(256)
void lstm_ew_kernel(const float* __restrict__ gates,
                    const float* __restrict__ cprime,
                    float* __restrict__ out)
{
    size_t idx = (size_t)blockIdx.x * 256 + threadIdx.x;   // Nn*Hc total
    int n = (int)(idx >> 9), hcol = (int)(idx & (Hc - 1));
    const float* g = gates + (size_t)n * G4;
    float ig = sigmoidf_(g[hcol]);
    float fg = sigmoidf_(g[Hc + hcol]);
    float gg = tanhf(g[2 * Hc + hcol]);
    float og = sigmoidf_(g[3 * Hc + hcol]);
    float cp = cprime[(size_t)n * Hc + hcol];
    float cn = fg * cp + ig * gg;
    float hn = og * tanhf(cn);
    out[(size_t)n * Hc + hcol] = hn;                          // h_new
    out[(size_t)Nn * Hc + (size_t)n * Hc + hcol] = cn;        // c_new
}

// ---------------------------------------------------------------------------
extern "C" void kernel_launch(void* const* d_in, const int* in_sizes, int n_in,
                              void* d_out, int out_size, void* d_ws, size_t ws_size,
                              hipStream_t stream) {
    (void)in_sizes; (void)n_in; (void)out_size; (void)ws_size;
    const float* x_in  = (const float*)d_in[0];
    const float* h     = (const float*)d_in[1];
    const float* c     = (const float*)d_in[2];
    const float* hs    = (const float*)d_in[3];
    const float* a_src = (const float*)d_in[4];
    const float* w_fg  = (const float*)d_in[5];
    const float* b_fg  = (const float*)d_in[6];
    const float* w_ig  = (const float*)d_in[7];
    const float* b_ig  = (const float*)d_in[8];
    const float* w_ih  = (const float*)d_in[9];
    const float* w_hh  = (const float*)d_in[10];
    const float* b_ih  = (const float*)d_in[11];
    const float* b_hh  = (const float*)d_in[12];
    float* out = (float*)d_out;

    // workspace layout
    char* ws = (char*)d_ws;
    _Float16* aggH = (_Float16*)ws;                 ws += (size_t)Nn * AGGK * 2;  // 32 MB
    _Float16* AgH  = (_Float16*)ws;                 ws += (size_t)Nn * KC * 2;    // 10 MB
    _Float16* WspH = (_Float16*)ws;                 ws += (size_t)1024 * AGGK * 2;// 4 MB
    _Float16* WgH  = (_Float16*)ws;                 ws += (size_t)G4 * KC * 2;    // 2.5 MB
    float*    cprime = (float*)ws;                  ws += (size_t)Nn * Hc * 4;    // 16 MB
    float*    gates  = (float*)ws;                                                // 64 MB

    const size_t smemG = (size_t)2 * (MT + NT) * PSTR * sizeof(_Float16); // 72 KB
    auto g0 = gemm_f16_kernel<0>;
    auto g1 = gemm_f16_kernel<1>;
    (void)hipFuncSetAttribute(reinterpret_cast<const void*>(g0),
                              hipFuncAttributeMaxDynamicSharedMemorySize, (int)smemG);
    (void)hipFuncSetAttribute(reinterpret_cast<const void*>(g1),
                              hipFuncAttributeMaxDynamicSharedMemorySize, (int)smemG);

    // pack f16 operands
    cvt_f16_kernel<<<(Hc * AGGK + 255) / 256, 256, 0, stream>>>(w_fg, WspH, Hc * AGGK);
    cvt_f16_kernel<<<(Hc * AGGK + 255) / 256, 256, 0, stream>>>(w_ig, WspH + (size_t)Hc * AGGK, Hc * AGGK);
    pack_wg_kernel<<<(G4 * KC + 255) / 256, 256, 0, stream>>>(w_ih, w_hh, WgH);
    pack_x_kernel<<<(Nn * INc + 255) / 256, 256, 0, stream>>>(x_in, AgH);

    // attention + aggregation -> aggH
    att_agg_kernel<<<Nn / AROWS, 256, 0, stream>>>(h, hs, a_src, aggH);

    // spatial gates: (8192 x 1024) = aggH (8192x2048) @ [w_fg;w_ig]^T
    dim3 gridS(Nn / MT, 1024 / NT);
    g0<<<gridS, 256, smemG, stream>>>(aggH, AGGK, WspH, AGGK, AGGK,
                                      b_fg, b_ig, h, c, cprime, AgH, nullptr);

    // LSTM gates: (8192 x 2048) = [x|h'] (8192x640) @ [w_ih|w_hh]^T
    dim3 gridG(Nn / MT, G4 / NT);
    g1<<<gridG, 256, smemG, stream>>>(AgH, KC, WgH, KC, KC,
                                      b_ih, b_hh, nullptr, nullptr, nullptr, nullptr, gates);

    // LSTM elementwise -> out
    lstm_ew_kernel<<<(Nn * Hc) / 256, 256, 0, stream>>>(gates, cprime, out);
}
